// SwinTransformerBlock_11605001634470
// MI455X (gfx1250) — matrix operand order repaired and checked by reading, not compiled, not run on hardware
//
#include <hip/hip_runtime.h>

typedef __bf16 bf16;
typedef __attribute__((ext_vector_type(16))) __bf16 v16bf;
typedef __attribute__((ext_vector_type(8)))  float   v8f;

#define WSZ   7
#define NTOK  49
#define NPAD  64
#define CDIM  192
#define NHEADS 6
#define HDIM  32
#define HID   768

union FragB16 { v16bf v; uint4 q[2]; };

// A fragment (16x32 bf16), source row-major [M][K] with leading dim lda.
__device__ __forceinline__ v16bf load_frag_a(const bf16* base, int lda, int m0, int k0, int lane) {
  int g = lane >> 4, r = lane & 15;
  const bf16* p = base + (m0 + r) * lda + k0 + 8 * g;
  FragB16 f;
  f.q[0] = *(const uint4*)p;
  f.q[1] = *(const uint4*)(p + 16);
  return f.v;
}

// B fragment (32x16 bf16), source stored N-major [N][K] with leading dim ldb.
__device__ __forceinline__ v16bf load_frag_b(const bf16* base, int ldb, int n0, int k0, int lane) {
  int g = lane >> 4, r = lane & 15;
  const bf16* p = base + (n0 + r) * ldb + k0 + 8 * g;
  FragB16 f;
  f.q[0] = *(const uint4*)p;
  f.q[1] = *(const uint4*)(p + 16);
  return f.v;
}

__device__ __forceinline__ v8f wmma_bf16(v16bf a, v16bf b, v8f c) {
  return __builtin_amdgcn_wmma_f32_16x16x32_bf16(false, a, false, b, (short)0, c, false, false);
}

__device__ __forceinline__ float wave_sum(float v) {
#pragma unroll
  for (int o = 16; o > 0; o >>= 1) v += __shfl_xor(v, o, 32);
  return v;
}
__device__ __forceinline__ float wave_max(float v) {
#pragma unroll
  for (int o = 16; o > 0; o >>= 1) v = fmaxf(v, __shfl_xor(v, o, 32));
  return v;
}

// store one 16x16 f32 D tile as bf16, row-major dst[m][c], with (x+bias)*scale
__device__ __forceinline__ void store_tile_rm(bf16* dst, int ld, int m0, int c0,
                                              v8f acc, float bias, float scale,
                                              int g, int r) {
#pragma unroll
  for (int i = 0; i < 8; i++)
    dst[(m0 + 8 * g + i) * ld + c0 + r] = (bf16)((acc[i] + bias) * scale);
}
// store one 16x16 f32 D tile as bf16, transposed dst[c][m]
__device__ __forceinline__ void store_tile_cm(bf16* dst, int ld, int m0, int c0,
                                              v8f acc, float bias, int g, int r) {
#pragma unroll
  for (int i = 0; i < 8; i++)
    dst[(c0 + r) * ld + (m0 + 8 * g + i)] = (bf16)(acc[i] + bias);
}

// ---------- weight convert + transpose: w[K][N] (f32) -> o[N][K] (bf16) ----------
__global__ void convert_tr(const float* __restrict__ w, bf16* __restrict__ o, int K, int Nn) {
  int idx = blockIdx.x * 256 + threadIdx.x;
  if (idx < K * Nn) {
    int n = idx / K;
    int k = idx - n * K;
    o[idx] = (bf16)w[(size_t)k * Nn + n];
  }
}

// ---------- attention kernel: one block per window ----------
__global__ __launch_bounds__(256) void attn_kernel(
    const float* __restrict__ x, const float* __restrict__ mask,
    const int* __restrict__ relidx, const float* __restrict__ ln1s,
    const float* __restrict__ ln1b, const float* __restrict__ qkvb,
    const float* __restrict__ rpb, const float* __restrict__ projb,
    const bf16* __restrict__ qkvWt, const bf16* __restrict__ projWt,
    float* __restrict__ xprime)
{
  __shared__ __align__(16) bf16  sY[NPAD * CDIM];   // LN1 output; reused as ctx
  __shared__ __align__(16) bf16  sQ[NPAD * CDIM];   // q * (1/sqrt(d)) pre-scaled
  __shared__ __align__(16) bf16  sK[NPAD * CDIM];
  __shared__ __align__(16) bf16  sVT[CDIM * NPAD];  // v transposed: [channel][token]
  __shared__ __align__(16) float sS[NPAD * NPAD];   // one head of raw scores
  __shared__ __align__(16) bf16  sP[NPAD * NPAD];   // one head of softmax probs

  const int lane = threadIdx.x & 31;
  const int wave = threadIdx.x >> 5;
  const int wid  = blockIdx.x;       // 0..2047
  const int b    = wid >> 6;
  const int w    = wid & 63;
  const int wr   = w >> 3, wc = w & 7;
  const int g = lane >> 4, r = lane & 15;

  // ---- Phase 1: LN1 + cyclic-shift gather -> sY (rows 49..63 zero) ----
  for (int n = wave; n < NPAD; n += 8) {
    if (n < NTOK) {
      int tr = n / WSZ, tc = n - tr * WSZ;
      int gh = (wr * WSZ + tr + 3) % 56;
      int gw = (wc * WSZ + tc + 3) % 56;
      const float* px = x + ((size_t)(b * 56 + gh) * 56 + gw) * CDIM;
      float v[6], s = 0.f, s2 = 0.f;
#pragma unroll
      for (int j = 0; j < 6; j++) { v[j] = px[lane + 32 * j]; s += v[j]; s2 += v[j] * v[j]; }
      s = wave_sum(s); s2 = wave_sum(s2);
      float mu  = s * (1.f / CDIM);
      float var = s2 * (1.f / CDIM) - mu * mu;
      float inv = rsqrtf(var + 1e-6f);
#pragma unroll
      for (int j = 0; j < 6; j++) {
        int c = lane + 32 * j;
        sY[n * CDIM + c] = (bf16)((v[j] - mu) * inv * ln1s[c] + ln1b[c]);
      }
    } else {
#pragma unroll
      for (int j = 0; j < 6; j++) sY[n * CDIM + lane + 32 * j] = (bf16)0.f;
    }
  }
  __syncthreads();

  // ---- Phase 2: QKV GEMM  sY(64x192) x Wqkv(192x576), 2x2 register blocking ----
  // M: 2 pairs of 16 (64), N: 18 pairs of 16 (576) -> 36 blocks
  for (int t = wave; t < 36; t += 8) {
    int mp = t / 18, np = t - mp * 18;
    int m0 = mp * 32, n0 = np * 32;
    v8f acc[2][2] = {};
#pragma unroll
    for (int k0 = 0; k0 < CDIM; k0 += 32) {
      v16bf a0 = load_frag_a(sY, CDIM, m0,      k0, lane);
      v16bf a1 = load_frag_a(sY, CDIM, m0 + 16, k0, lane);
      v16bf b0 = load_frag_b(qkvWt, CDIM, n0,      k0, lane);
      v16bf b1 = load_frag_b(qkvWt, CDIM, n0 + 16, k0, lane);
      acc[0][0] = wmma_bf16(a0, b0, acc[0][0]);
      acc[0][1] = wmma_bf16(a0, b1, acc[0][1]);
      acc[1][0] = wmma_bf16(a1, b0, acc[1][0]);
      acc[1][1] = wmma_bf16(a1, b1, acc[1][1]);
    }
    float bias0 = qkvb[n0 + r];
    float bias1 = qkvb[n0 + 16 + r];
    // region is uniform per block: [0,192)=q, [192,384)=k, [384,576)=v
    if (n0 < CDIM) {            // q: pre-scale by 1/sqrt(HDIM)
      const float sc = 0.17677669529663687f;
      store_tile_rm(sQ, CDIM, m0,      n0,      acc[0][0], bias0, sc, g, r);
      store_tile_rm(sQ, CDIM, m0,      n0 + 16, acc[0][1], bias1, sc, g, r);
      store_tile_rm(sQ, CDIM, m0 + 16, n0,      acc[1][0], bias0, sc, g, r);
      store_tile_rm(sQ, CDIM, m0 + 16, n0 + 16, acc[1][1], bias1, sc, g, r);
    } else if (n0 < 2 * CDIM) { // k
      int c0 = n0 - CDIM;
      store_tile_rm(sK, CDIM, m0,      c0,      acc[0][0], bias0, 1.f, g, r);
      store_tile_rm(sK, CDIM, m0,      c0 + 16, acc[0][1], bias1, 1.f, g, r);
      store_tile_rm(sK, CDIM, m0 + 16, c0,      acc[1][0], bias0, 1.f, g, r);
      store_tile_rm(sK, CDIM, m0 + 16, c0 + 16, acc[1][1], bias1, 1.f, g, r);
    } else {                    // v, transposed [channel][token]
      int c0 = n0 - 2 * CDIM;
      store_tile_cm(sVT, NPAD, m0,      c0,      acc[0][0], bias0, g, r);
      store_tile_cm(sVT, NPAD, m0,      c0 + 16, acc[0][1], bias1, g, r);
      store_tile_cm(sVT, NPAD, m0 + 16, c0,      acc[1][0], bias0, g, r);
      store_tile_cm(sVT, NPAD, m0 + 16, c0 + 16, acc[1][1], bias1, g, r);
    }
  }
  __syncthreads();

  // ---- Phase 3: per-head attention ----
  for (int h = 0; h < NHEADS; h++) {
    // scores = (q*sc) @ k^T  (K = HDIM = 32 -> single chained WMMA), 1x2 blocking
    for (int t = wave; t < 8; t += 8) {
      int m0 = (t >> 1) * 16, n0 = (t & 1) * 32;
      v16bf a  = load_frag_a(sQ + h * HDIM, CDIM, m0, 0, lane);
      v16bf b0 = load_frag_b(sK + h * HDIM, CDIM, n0,      0, lane);
      v16bf b1 = load_frag_b(sK + h * HDIM, CDIM, n0 + 16, 0, lane);
      v8f acc0 = {}, acc1 = {};
      acc0 = wmma_bf16(a, b0, acc0);
      acc1 = wmma_bf16(a, b1, acc1);
#pragma unroll
      for (int i = 0; i < 8; i++) {
        sS[(m0 + 8 * g + i) * NPAD + n0 + r]      = acc0[i];
        sS[(m0 + 8 * g + i) * NPAD + n0 + 16 + r] = acc1[i];
      }
    }
    __syncthreads();

    // softmax rows (relative position bias + shift mask; scale already in q)
    for (int n = wave; n < NPAD; n += 8) {
      if (n < NTOK) {
        int m0i = lane, m1i = lane + 32;
        float l0 = sS[n * NPAD + m0i]
                 + rpb[relidx[n * NTOK + m0i] * NHEADS + h]
                 + mask[((size_t)w * NTOK + n) * NTOK + m0i];
        float l1 = -1e30f;
        if (m1i < NTOK)
          l1 = sS[n * NPAD + m1i]
             + rpb[relidx[n * NTOK + m1i] * NHEADS + h]
             + mask[((size_t)w * NTOK + n) * NTOK + m1i];
        float mx = wave_max(fmaxf(l0, l1));
        float e0 = __expf(l0 - mx);
        float e1 = (m1i < NTOK) ? __expf(l1 - mx) : 0.f;
        float rs = 1.f / wave_sum(e0 + e1);
        sP[n * NPAD + m0i] = (bf16)(e0 * rs);
        sP[n * NPAD + m1i] = (bf16)(e1 * rs);
      } else {
        sP[n * NPAD + lane]      = (bf16)0.f;
        sP[n * NPAD + lane + 32] = (bf16)0.f;
      }
    }
    __syncthreads();

    // ctx = probs(64x64) @ v(64x32) -> bf16 into sY[token][h*32 + d]
    for (int t = wave; t < 8; t += 8) {
      int m0 = (t >> 1) * 16, n0 = (t & 1) * 16;
      v8f acc = {};
#pragma unroll
      for (int k0 = 0; k0 < NPAD; k0 += 32) {
        v16bf a  = load_frag_a(sP, NPAD, m0, k0, lane);
        v16bf bb = load_frag_b(sVT + h * HDIM * NPAD, NPAD, n0, k0, lane);
        acc = wmma_bf16(a, bb, acc);
      }
#pragma unroll
      for (int i = 0; i < 8; i++)
        sY[(m0 + 8 * g + i) * CDIM + h * HDIM + n0 + r] = (bf16)acc[i];
    }
    __syncthreads();
  }

  // ---- Phase 4: proj GEMM (2x2 blocking) + un-shift scatter + residual ----
  // M: 2 pairs, N: 6 pairs -> 12 blocks
  for (int t = wave; t < 12; t += 8) {
    int mp = t / 6, np = t - mp * 6;
    int m0 = mp * 32, n0 = np * 32;
    v8f acc[2][2] = {};
#pragma unroll
    for (int k0 = 0; k0 < CDIM; k0 += 32) {
      v16bf a0 = load_frag_a(sY, CDIM, m0,      k0, lane);
      v16bf a1 = load_frag_a(sY, CDIM, m0 + 16, k0, lane);
      v16bf b0 = load_frag_b(projWt, CDIM, n0,      k0, lane);
      v16bf b1 = load_frag_b(projWt, CDIM, n0 + 16, k0, lane);
      acc[0][0] = wmma_bf16(a0, b0, acc[0][0]);
      acc[0][1] = wmma_bf16(a0, b1, acc[0][1]);
      acc[1][0] = wmma_bf16(a1, b0, acc[1][0]);
      acc[1][1] = wmma_bf16(a1, b1, acc[1][1]);
    }
#pragma unroll
    for (int mi = 0; mi < 2; mi++) {
#pragma unroll
      for (int ni = 0; ni < 2; ni++) {
        int c = n0 + ni * 16 + r;
        float bias = projb[c];
#pragma unroll
        for (int i = 0; i < 8; i++) {
          int n = m0 + mi * 16 + 8 * g + i;
          if (n < NTOK) {
            int tr = n / WSZ, tc = n - tr * WSZ;
            int gh = (wr * WSZ + tr + 3) % 56;
            int gw = (wc * WSZ + tc + 3) % 56;
            size_t off = ((size_t)(b * 56 + gh) * 56 + gw) * CDIM + c;
            xprime[off] = x[off] + acc[mi][ni][i] + bias;
          }
        }
      }
    }
  }
}

// ---------- MLP kernel: 64-token tiles ----------
__global__ __launch_bounds__(256) void mlp_kernel(
    const float* __restrict__ xprime, const float* __restrict__ ln2s,
    const float* __restrict__ ln2b, const float* __restrict__ b1,
    const float* __restrict__ b2, const bf16* __restrict__ w1t,
    const bf16* __restrict__ w2t, float* __restrict__ out)
{
  __shared__ __align__(16) bf16 sA[64 * CDIM];
  __shared__ __align__(16) bf16 sH[64 * HID];

  const int lane = threadIdx.x & 31;
  const int wave = threadIdx.x >> 5;
  const int g = lane >> 4, r = lane & 15;
  const size_t t0 = (size_t)blockIdx.x * 64;

  // LN2 -> sA
  for (int n = wave; n < 64; n += 8) {
    const float* px = xprime + (t0 + n) * CDIM;
    float v[6], s = 0.f, s2 = 0.f;
#pragma unroll
    for (int j = 0; j < 6; j++) { v[j] = px[lane + 32 * j]; s += v[j]; s2 += v[j] * v[j]; }
    s = wave_sum(s); s2 = wave_sum(s2);
    float mu  = s * (1.f / CDIM);
    float var = s2 * (1.f / CDIM) - mu * mu;
    float inv = rsqrtf(var + 1e-6f);
#pragma unroll
    for (int j = 0; j < 6; j++) {
      int c = lane + 32 * j;
      sA[n * CDIM + c] = (bf16)((v[j] - mu) * inv * ln2s[c] + ln2b[c]);
    }
  }
  __syncthreads();

  // GEMM1 (192 -> 768) + tanh-GELU -> sH ; 2x2 blocking: 2 x 24 = 48 blocks
  for (int t = wave; t < 48; t += 8) {
    int mp = t / 24, np = t - mp * 24;
    int m0 = mp * 32, n0 = np * 32;
    v8f acc[2][2] = {};
#pragma unroll
    for (int k0 = 0; k0 < CDIM; k0 += 32) {
      v16bf a0 = load_frag_a(sA, CDIM, m0,      k0, lane);
      v16bf a1 = load_frag_a(sA, CDIM, m0 + 16, k0, lane);
      v16bf b0 = load_frag_b(w1t, CDIM, n0,      k0, lane);
      v16bf b1 = load_frag_b(w1t, CDIM, n0 + 16, k0, lane);
      acc[0][0] = wmma_bf16(a0, b0, acc[0][0]);
      acc[0][1] = wmma_bf16(a0, b1, acc[0][1]);
      acc[1][0] = wmma_bf16(a1, b0, acc[1][0]);
      acc[1][1] = wmma_bf16(a1, b1, acc[1][1]);
    }
#pragma unroll
    for (int mi = 0; mi < 2; mi++) {
#pragma unroll
      for (int ni = 0; ni < 2; ni++) {
        int c = n0 + ni * 16 + r;
        float bias = b1[c];
#pragma unroll
        for (int i = 0; i < 8; i++) {
          float z  = acc[mi][ni][i] + bias;
          float gz = 0.5f * z * (1.f + tanhf(0.7978845608f * (z + 0.044715f * z * z * z)));
          sH[(m0 + mi * 16 + 8 * g + i) * HID + c] = (bf16)gz;
        }
      }
    }
  }
  __syncthreads();

  // GEMM2 (768 -> 192) + residual -> out ; 2x2 blocking: 2 x 6 = 12 blocks
  for (int t = wave; t < 12; t += 8) {
    int mp = t / 6, np = t - mp * 6;
    int m0 = mp * 32, n0 = np * 32;
    v8f acc[2][2] = {};
#pragma unroll
    for (int k0 = 0; k0 < HID; k0 += 32) {
      v16bf a0 = load_frag_a(sH, HID, m0,      k0, lane);
      v16bf a1 = load_frag_a(sH, HID, m0 + 16, k0, lane);
      v16bf b0 = load_frag_b(w2t, HID, n0,      k0, lane);
      v16bf b1 = load_frag_b(w2t, HID, n0 + 16, k0, lane);
      acc[0][0] = wmma_bf16(a0, b0, acc[0][0]);
      acc[0][1] = wmma_bf16(a0, b1, acc[0][1]);
      acc[1][0] = wmma_bf16(a1, b0, acc[1][0]);
      acc[1][1] = wmma_bf16(a1, b1, acc[1][1]);
    }
#pragma unroll
    for (int mi = 0; mi < 2; mi++) {
#pragma unroll
      for (int ni = 0; ni < 2; ni++) {
        int c = n0 + ni * 16 + r;
        float bias = b2[c];
#pragma unroll
        for (int i = 0; i < 8; i++) {
          size_t off = (t0 + m0 + mi * 16 + 8 * g + i) * CDIM + c;
          out[off] = xprime[off] + acc[mi][ni][i] + bias;
        }
      }
    }
  }
}

extern "C" void kernel_launch(void* const* d_in, const int* in_sizes, int n_in,
                              void* d_out, int out_size, void* d_ws, size_t ws_size,
                              hipStream_t stream) {
  const float* x      = (const float*)d_in[0];
  const float* mask   = (const float*)d_in[1];
  const int*   relidx = (const int*)d_in[2];
  const float* ln1s   = (const float*)d_in[3];
  const float* ln1b   = (const float*)d_in[4];
  const float* qkvw   = (const float*)d_in[5];
  const float* qkvb   = (const float*)d_in[6];
  const float* rpb    = (const float*)d_in[7];
  const float* projw  = (const float*)d_in[8];
  const float* projb  = (const float*)d_in[9];
  const float* ln2s   = (const float*)d_in[10];
  const float* ln2b   = (const float*)d_in[11];
  const float* w1     = (const float*)d_in[12];
  const float* b1     = (const float*)d_in[13];
  const float* w2     = (const float*)d_in[14];
  const float* b2     = (const float*)d_in[15];

  char* ws = (char*)d_ws;
  bf16* qkvWt  = (bf16*)ws;                 // [576][192]
  bf16* projWt = qkvWt  + 576 * 192;        // [192][192]
  bf16* w1t    = projWt + 192 * 192;        // [768][192]
  bf16* w2t    = w1t    + 768 * 192;        // [192][768]
  float* xprime = (float*)(ws + (1 << 20)); // [100352][192] f32

  convert_tr<<<(192 * 576 + 255) / 256, 256, 0, stream>>>(qkvw, qkvWt, 192, 576);
  convert_tr<<<(192 * 192 + 255) / 256, 256, 0, stream>>>(projw, projWt, 192, 192);
  convert_tr<<<(192 * 768 + 255) / 256, 256, 0, stream>>>(w1, w1t, 192, 768);
  convert_tr<<<(768 * 192 + 255) / 256, 256, 0, stream>>>(w2, w2t, 768, 192);

  attn_kernel<<<2048, 256, 0, stream>>>(x, mask, relidx, ln1s, ln1b, qkvb, rpb, projb,
                                        qkvWt, projWt, xprime);
  mlp_kernel<<<1568, 256, 0, stream>>>(xprime, ln2s, ln2b, b1, b2, w1t, w2t, (float*)d_out);
}
